// TriadTransformer_53489522704517
// MI455X (gfx1250) — matrix operand order repaired and checked by reading, not compiled, not run on hardware
//
#include <hip/hip_runtime.h>
#include <hip/hip_bf16.h>
#include <math.h>

typedef unsigned short u16;
typedef __bf16 bf16_t;
typedef __attribute__((ext_vector_type(16))) __bf16          bf16x16;
typedef __attribute__((ext_vector_type(8)))  float           v8f;
typedef __attribute__((ext_vector_type(8)))  unsigned short  ushort8;

constexpr int Bq = 8, Sq = 512, Dq = 512, Hq = 8, Fq = 2048, Lq = 6, Vq = 32000;
constexpr int HDq = Dq / Hq;      // 64
constexpr int BSq = Bq * Sq;      // 4096

#if defined(__has_builtin)
#if __has_builtin(__builtin_amdgcn_global_load_async_to_lds_b128)
#define HAS_ASYNC_LDS 1
#endif
#endif
#ifndef HAS_ASYNC_LDS
#define HAS_ASYNC_LDS 0
#endif

__device__ __forceinline__ u16 f2bf(float f) {
  union { float f; unsigned u; } c; c.f = f;
  return (u16)((c.u + 0x7FFFu + ((c.u >> 16) & 1u)) >> 16);
}

union FragBF { bf16x16 v; ushort8 h[2]; };

#if HAS_ASYNC_LDS
// builtin signature (from probe diagnostics): param1 = v4i in device AS(1),
// param2 = LDS side, params 3/4 = imm offset / cpol
typedef int v4i_gcc __attribute__((vector_size(16)));
typedef __attribute__((address_space(1))) v4i_gcc* as1_v4i;
typedef __attribute__((address_space(3))) v4i_gcc* as3_v4i;
__device__ __forceinline__ void async_copy16(const u16* g, u16* l) {
  // global_load_async_to_lds_b128: 16B per lane, tracked by ASYNCcnt
  __builtin_amdgcn_global_load_async_to_lds_b128(
      (as1_v4i)(v4i_gcc*)(void*)g, (as3_v4i)(v4i_gcc*)(void*)l, 0, 0);
}
#endif

// ---------------------------------------------------------------------------
// Tiled WMMA GEMM:  C[M,N] = act(alpha * A[M,K] @ B[N,K]^T + bias[N])
// A, B bf16 row-major (both walked along K), C fp32 and/or bf16.
// Batched via blockIdx.z:  off = (z/zdiv)*s1 + (z%zdiv)*s2 per operand.
// 8 wave32 waves; per-wave TMxTN grid of 16x16 tiles (v_wmma_f32_16x16x32_bf16).
// Double-buffered LDS; next tile staged (async-to-LDS if available, else via
// registers with deferred loadcnt waits) while WMMAs run on the current tile.
// ---------------------------------------------------------------------------
template<int BM, int BN, int TM, int TN>
__global__ __launch_bounds__(256)
void gemm_bf16_nt(const u16* __restrict__ A, const u16* __restrict__ Bw,
                  const float* __restrict__ bias,
                  float* __restrict__ C, u16* __restrict__ Cb,
                  int M, int N, int K, int lda, int ldb, int ldc,
                  int zdiv,
                  long long sA1, long long sA2,
                  long long sB1, long long sB2,
                  long long sC1, long long sC2,
                  float alpha, int relu)
{
  constexpr int KT   = 32;
  constexpr int LDSK = KT + 8;               // 80B rows: keeps 16B alignment
  constexpr int WMW  = BM / (16 * TM);
  constexpr int WNW  = BN / (16 * TN);
  constexpr int ACH  = (BM * 4) / 256;       // 16B chunks per thread for A
  constexpr int BCH  = (BN * 4) / 256;       // 16B chunks per thread for B
  static_assert(WMW * WNW == 8, "need exactly 8 waves");
  static_assert(ACH >= 1 && BCH >= 1, "tile too small for 256 threads");

  __shared__ u16 lA[2][BM * LDSK];
  __shared__ u16 lB[2][BN * LDSK];

  const int z = blockIdx.z;
  const u16* Ab = A  + (long long)(z / zdiv) * sA1 + (long long)(z % zdiv) * sA2;
  const u16* Bb = Bw + (long long)(z / zdiv) * sB1 + (long long)(z % zdiv) * sB2;
  const long long offC = (long long)(z / zdiv) * sC1 + (long long)(z % zdiv) * sC2;

  const int bm0  = blockIdx.y * BM;
  const int bn0  = blockIdx.x * BN;
  const int tid  = threadIdx.x;
  const int lane = tid & 31;
  const int wave = tid >> 5;
  const int wm   = wave % WMW;
  const int wn   = wave / WMW;
  const int mrow = lane & 15;
  const int khalf = lane >> 4;

  const v8f vzero = {0.f,0.f,0.f,0.f,0.f,0.f,0.f,0.f};
  v8f acc[TM][TN];
#pragma unroll
  for (int a = 0; a < TM; ++a)
#pragma unroll
    for (int b = 0; b < TN; ++b) acc[a][b] = vzero;

  // per-thread chunk coordinates (fixed -> no exec-mask branching)
  int arow[ACH], acol[ACH], brow[BCH], bcol[BCH];
#pragma unroll
  for (int i = 0; i < ACH; ++i) { const int idx = tid + i * 256; arow[i] = idx >> 2; acol[i] = (idx & 3) * 8; }
#pragma unroll
  for (int i = 0; i < BCH; ++i) { const int idx = tid + i * 256; brow[i] = idx >> 2; bcol[i] = (idx & 3) * 8; }

#if HAS_ASYNC_LDS
  // ---------------- async-to-LDS double-buffered pipeline -----------------
  auto stage = [&](int buf, int k0) {
#pragma unroll
    for (int i = 0; i < ACH; ++i)
      async_copy16(&Ab[(long long)(bm0 + arow[i]) * lda + k0 + acol[i]],
                   &lA[buf][arow[i] * LDSK + acol[i]]);
#pragma unroll
    for (int i = 0; i < BCH; ++i)
      async_copy16(&Bb[(long long)(bn0 + brow[i]) * ldb + k0 + bcol[i]],
                   &lB[buf][brow[i] * LDSK + bcol[i]]);
  };
  stage(0, 0);
#else
  // ---------------- register-staged double-buffered pipeline --------------
  ushort8 ra[ACH], rb[BCH];
  auto loadRegs = [&](int k0) {
#pragma unroll
    for (int i = 0; i < ACH; ++i)
      ra[i] = *(const ushort8*)&Ab[(long long)(bm0 + arow[i]) * lda + k0 + acol[i]];
#pragma unroll
    for (int i = 0; i < BCH; ++i)
      rb[i] = *(const ushort8*)&Bb[(long long)(bn0 + brow[i]) * ldb + k0 + bcol[i]];
  };
  auto storeRegs = [&](int buf) {
#pragma unroll
    for (int i = 0; i < ACH; ++i) *(ushort8*)&lA[buf][arow[i] * LDSK + acol[i]] = ra[i];
#pragma unroll
    for (int i = 0; i < BCH; ++i) *(ushort8*)&lB[buf][brow[i] * LDSK + bcol[i]] = rb[i];
  };
  loadRegs(0);
#endif

  int it = 0;
  for (int k0 = 0; k0 < K; k0 += KT, ++it) {
    const int buf = it & 1;
#if HAS_ASYNC_LDS
    asm volatile("s_wait_asynccnt 0" ::: "memory");  // this buffer's DMA done
    __syncthreads();                                 // visible to all waves
    if (k0 + KT < K) stage(buf ^ 1, k0 + KT);        // overlap next tile DMA
#else
    storeRegs(buf);                                  // waits its own loadcnt
    __syncthreads();
    if (k0 + KT < K) loadRegs(k0 + KT);              // loads fly over compute
#endif

    // ---- fragments per the 16-bit A 16x32 / B 32x16 lane layouts
    FragBF af[TM], bfw[TN];
#pragma unroll
    for (int tm = 0; tm < TM; ++tm) {
      const u16* p = &lA[buf][((wm * TM + tm) * 16 + mrow) * LDSK + khalf * 8];
      af[tm].h[0] = *(const ushort8*)p;         // K = kb .. kb+7
      af[tm].h[1] = *(const ushort8*)(p + 16);  // K = kb+16 .. kb+23
    }
#pragma unroll
    for (int tn = 0; tn < TN; ++tn) {
      const u16* p = &lB[buf][((wn * TN + tn) * 16 + mrow) * LDSK + khalf * 8];
      bfw[tn].h[0] = *(const ushort8*)p;         // K = kb .. kb+7
      bfw[tn].h[1] = *(const ushort8*)(p + 16);  // K = kb+16 .. kb+23
    }
#pragma unroll
    for (int tm = 0; tm < TM; ++tm)
#pragma unroll
      for (int tn = 0; tn < TN; ++tn)
        acc[tm][tn] = __builtin_amdgcn_wmma_f32_16x16x32_bf16(
            false, af[tm].v, false, bfw[tn].v, (short)0, acc[tm][tn],
            false, false);
  }

  // ---- epilogue: alpha, bias, relu; fp32 and/or bf16 stores
#pragma unroll
  for (int tm = 0; tm < TM; ++tm) {
#pragma unroll
    for (int tn = 0; tn < TN; ++tn) {
      const int nl = bn0 + (wn * TN + tn) * 16 + mrow;
      const float bv = bias ? bias[nl] : 0.f;
#pragma unroll
      for (int r = 0; r < 8; ++r) {
        const int ml = bm0 + (wm * TM + tm) * 16 + khalf * 8 + r;
        float v = acc[tm][tn][r] * alpha + bv;
        if (relu) v = fmaxf(v, 0.f);
        const long long ci = offC + (long long)ml * ldc + nl;
        if (C)  C[ci]  = v;
        if (Cb) Cb[ci] = f2bf(v);
      }
    }
  }
}

// ---------------------------------------------------------------------------
__global__ __launch_bounds__(256)
void f32_to_bf16_kernel(const float* __restrict__ in, u16* __restrict__ out,
                        long long n) {
  long long i = (long long)blockIdx.x * 256 + threadIdx.x;
  const long long stride = (long long)gridDim.x * 256;
  for (; i < n; i += stride) out[i] = f2bf(in[i]);
}

__global__ __launch_bounds__(256)
void embed_kernel(const int* __restrict__ x, const float* __restrict__ emb,
                  const float* __restrict__ pos,
                  float* __restrict__ h, u16* __restrict__ hb) {
  const int i = blockIdx.x * 256 + threadIdx.x;
  if (i >= BSq * Dq) return;
  const int d   = i & (Dq - 1);
  const int row = i >> 9;              // / 512
  const int s   = row & (Sq - 1);
  const float v = emb[(long long)x[row] * Dq + d] + pos[s * Dq + d];
  h[i] = v; hb[i] = f2bf(v);
}

// vt[b,h,d,s] = qkv[b,s, 2D + h*HD + d]
__global__ __launch_bounds__(256)
void vt_transpose(const u16* __restrict__ qkv, u16* __restrict__ vt) {
  const int idx = blockIdx.x * 256 + threadIdx.x;
  if (idx >= Bq * Dq * Sq) return;
  const int s = idx & (Sq - 1);
  int t = idx >> 9;
  const int d = t % HDq; t /= HDq;
  const int h = t % Hq;
  const int b = t / Hq;
  vt[idx] = qkv[((long long)(b * Sq + s)) * (3 * Dq) + 2 * Dq + h * HDq + d];
}

__global__ __launch_bounds__(128)
void softmax_rows(const float* __restrict__ sc, u16* __restrict__ pr) {
  __shared__ float red[128];
  const long long rowoff = (long long)blockIdx.x * Sq;
  const int tid = threadIdx.x;
  float v[4];
  float m = -3.4e38f;
#pragma unroll
  for (int i = 0; i < 4; ++i) { v[i] = sc[rowoff + tid + i * 128]; m = fmaxf(m, v[i]); }
  red[tid] = m; __syncthreads();
  for (int s2 = 64; s2 > 0; s2 >>= 1) { if (tid < s2) red[tid] = fmaxf(red[tid], red[tid + s2]); __syncthreads(); }
  m = red[0]; __syncthreads();
  float sum = 0.f;
#pragma unroll
  for (int i = 0; i < 4; ++i) { v[i] = __expf(v[i] - m); sum += v[i]; }
  red[tid] = sum; __syncthreads();
  for (int s2 = 64; s2 > 0; s2 >>= 1) { if (tid < s2) red[tid] += red[tid + s2]; __syncthreads(); }
  const float inv = 1.f / red[0];
#pragma unroll
  for (int i = 0; i < 4; ++i) pr[rowoff + tid + i * 128] = f2bf(v[i] * inv);
}

__global__ __launch_bounds__(128)
void ln_residual(const float* __restrict__ hin, const float* __restrict__ addv,
                 const float* __restrict__ gam, const float* __restrict__ bet,
                 float* __restrict__ hout, u16* __restrict__ hbout) {
  __shared__ float r1[128], r2[128];
  const long long rowoff = (long long)blockIdx.x * Dq;
  const int tid = threadIdx.x;
  float v[4]; float s = 0.f, ss = 0.f;
#pragma unroll
  for (int i = 0; i < 4; ++i) {
    const int d = tid + i * 128;
    v[i] = hin[rowoff + d] + addv[rowoff + d];
    s += v[i]; ss += v[i] * v[i];
  }
  r1[tid] = s; r2[tid] = ss; __syncthreads();
  for (int st = 64; st > 0; st >>= 1) {
    if (tid < st) { r1[tid] += r1[tid + st]; r2[tid] += r2[tid + st]; }
    __syncthreads();
  }
  const float mean = r1[0] / Dq;
  const float var  = r2[0] / Dq - mean * mean;
  const float inv  = rsqrtf(var + 1e-5f);
#pragma unroll
  for (int i = 0; i < 4; ++i) {
    const int d = tid + i * 128;
    const float y = gam[d] * (v[i] - mean) * inv + bet[d];
    hout[rowoff + d] = y; hbout[rowoff + d] = f2bf(y);
  }
}

// ---------------- triad cycle pieces --------------------------------------
__global__ void zero_f32(float* p, int n) {
  const int i = blockIdx.x * 256 + threadIdx.x;
  if (i < n) p[i] = 0.f;
}

__global__ __launch_bounds__(256)
void sqnorm_partial(const float* __restrict__ h, float* __restrict__ out) {
  __shared__ float red[256];
  long long i = (long long)blockIdx.x * 256 + threadIdx.x;
  const long long n = (long long)BSq * Dq;
  const long long stride = (long long)gridDim.x * 256;
  float s = 0.f;
  for (; i < n; i += stride) { const float v = h[i]; s += v * v; }
  red[threadIdx.x] = s; __syncthreads();
  for (int st = 128; st > 0; st >>= 1) {
    if ((int)threadIdx.x < st) red[threadIdx.x] += red[threadIdx.x + st];
    __syncthreads();
  }
  if (threadIdx.x == 0) atomicAdd(out, red[0]);
}

__global__ __launch_bounds__(128)
void row_g_kernel(const float* __restrict__ h, const float* __restrict__ sq,
                  float* __restrict__ g) {
  __shared__ float red[128];
  const long long rowoff = (long long)blockIdx.x * Dq;
  float s = 0.f;
#pragma unroll
  for (int i = 0; i < 4; ++i) s += h[rowoff + threadIdx.x + i * 128];
  red[threadIdx.x] = s; __syncthreads();
  for (int st = 64; st > 0; st >>= 1) {
    if ((int)threadIdx.x < st) red[threadIdx.x] += red[threadIdx.x + st];
    __syncthreads();
  }
  if (threadIdx.x == 0) g[blockIdx.x] = (red[0] / Dq) / (sqrtf(*sq) + 1e-8f);
}

// one-block bitonic sort of |g| (4096), quantile(0.75) threshold, mod row
__global__ __launch_bounds__(1024)
void quantile_mod(const float* __restrict__ g, float* __restrict__ modrow,
                  float damp) {
  __shared__ float s[BSq];
  const int tid = threadIdx.x;
  for (int i = tid; i < BSq; i += 1024) s[i] = fabsf(g[i]);
  __syncthreads();
  for (int k = 2; k <= BSq; k <<= 1) {
    for (int j = k >> 1; j > 0; j >>= 1) {
      for (int i = tid; i < BSq; i += 1024) {
        const int ixj = i ^ j;
        if (ixj > i) {
          const float a = s[i], b = s[ixj];
          const bool up = ((i & k) == 0);
          if ((a > b) == up) { s[i] = b; s[ixj] = a; }
        }
      }
      __syncthreads();
    }
  }
  __shared__ float thrS;
  if (tid == 0) {
    // quantile pos = 0.75*(4096-1) = 3071.25 -> lerp
    const float a = s[3071], b = s[3072];
    thrS = a + 0.25f * (b - a);
  }
  __syncthreads();
  const float thr = thrS;
  for (int c = tid; c < Dq; c += 1024) {
    const float gv = g[c];                  // tile(mod)[:D] == mod[:D]
    modrow[c] = (fabsf(gv) > thr) ? gv * damp : 0.f;
  }
}

__global__ __launch_bounds__(256)
void triad_add(float* __restrict__ h, u16* __restrict__ hb,
               const float* __restrict__ modrow) {
  const int i = blockIdx.x * 256 + threadIdx.x;
  if (i >= BSq * Dq) return;
  const float v = h[i] + modrow[i & (Dq - 1)];
  h[i] = v; hb[i] = f2bf(v);
}

// ---------------------------------------------------------------------------
extern "C" void kernel_launch(void* const* d_in, const int* in_sizes, int n_in,
                              void* d_out, int out_size, void* d_ws, size_t ws_size,
                              hipStream_t stream) {
  (void)in_sizes; (void)n_in; (void)out_size; (void)ws_size;
  const int*   x         = (const int*)  d_in[0];
  const float* emb       = (const float*)d_in[1];
  const float* pos       = (const float*)d_in[2];
  const float* in_proj_w = (const float*)d_in[3];
  const float* in_proj_b = (const float*)d_in[4];
  const float* out_w     = (const float*)d_in[5];
  const float* out_b     = (const float*)d_in[6];
  const float* ln1_g     = (const float*)d_in[7];
  const float* ln1_b     = (const float*)d_in[8];
  const float* ln2_g     = (const float*)d_in[9];
  const float* ln2_b     = (const float*)d_in[10];
  const float* ff1_w     = (const float*)d_in[11];
  const float* ff1_b     = (const float*)d_in[12];
  const float* ff2_w     = (const float*)d_in[13];
  const float* ff2_b     = (const float*)d_in[14];
  const float* Wout      = (const float*)d_in[15];
  const float* bout      = (const float*)d_in[16];
  float* logits = (float*)d_out;

  char* base = (char*)d_ws; size_t off = 0;
  auto alloc = [&](size_t bytes) -> void* {
    off = (off + 255) & ~(size_t)255;
    void* p = base + off; off += bytes; return p;
  };
  float* h        = (float*)alloc((size_t)BSq * Dq * 4);
  u16*   hb       = (u16*)  alloc((size_t)BSq * Dq * 2);
  u16*   qkv_bf   = (u16*)  alloc((size_t)BSq * 3 * Dq * 2);
  u16*   vt_bf    = (u16*)  alloc((size_t)BSq * Dq * 2);
  float* scores   = (float*)alloc((size_t)Bq * Hq * Sq * Sq * 4);
  u16*   probs_bf = (u16*)  alloc((size_t)Bq * Hq * Sq * Sq * 2);
  u16*   attn_bf  = (u16*)  alloc((size_t)BSq * Dq * 2);
  float* tmp      = (float*)alloc((size_t)BSq * Dq * 4);
  u16*   ff1_bf   = (u16*)  alloc((size_t)BSq * Fq * 2);
  u16*   w_inproj = (u16*)  alloc((size_t)Lq * 3 * Dq * Dq * 2);
  u16*   w_outw   = (u16*)  alloc((size_t)Lq * Dq * Dq * 2);
  u16*   w_ff1    = (u16*)  alloc((size_t)Lq * Fq * Dq * 2);
  u16*   w_ff2    = (u16*)  alloc((size_t)Lq * Dq * Fq * 2);
  u16*   w_vout   = (u16*)  alloc((size_t)Vq * Dq * 2);
  float* sqn      = (float*)alloc(256);
  float* gbuf     = (float*)alloc((size_t)BSq * 4);
  float* modrow   = (float*)alloc((size_t)Dq * 4);

  const float damp = (float)pow(0.99, 1.0 / (double)Lq);

  // ---- weights fp32 -> bf16 (deterministic, every call)
  f32_to_bf16_kernel<<<2048, 256, 0, stream>>>(in_proj_w, w_inproj, (long long)Lq * 3 * Dq * Dq);
  f32_to_bf16_kernel<<<2048, 256, 0, stream>>>(out_w,     w_outw,   (long long)Lq * Dq * Dq);
  f32_to_bf16_kernel<<<2048, 256, 0, stream>>>(ff1_w,     w_ff1,    (long long)Lq * Fq * Dq);
  f32_to_bf16_kernel<<<2048, 256, 0, stream>>>(ff2_w,     w_ff2,    (long long)Lq * Dq * Fq);
  f32_to_bf16_kernel<<<4096, 256, 0, stream>>>(Wout,      w_vout,   (long long)Vq * Dq);

  // ---- embedding + positional
  embed_kernel<<<(BSq * Dq) / 256, 256, 0, stream>>>(x, emb, pos, h, hb);

  const float iscale = 1.0f / sqrtf((float)HDq);

  for (int i = 0; i < Lq; ++i) {
    // qkv = h @ Wqkv^T + b         (4096 x 1536, K=512) -> bf16
    gemm_bf16_nt<128,128,2,4><<<dim3(3 * Dq / 128, BSq / 128, 1), 256, 0, stream>>>(
        hb, w_inproj + (size_t)i * 3 * Dq * Dq, in_proj_b + (size_t)i * 3 * Dq,
        nullptr, qkv_bf, BSq, 3 * Dq, Dq, Dq, Dq, 3 * Dq,
        1, 0, 0, 0, 0, 0, 0, 1.0f, 0);

    // V^T per head: vt[b,h,d,s]
    vt_transpose<<<(BSq * Dq) / 256, 256, 0, stream>>>(qkv_bf, vt_bf);

    // scores = (Q @ K^T) * 1/sqrt(HD)   per (b,h): 512x512, K=64
    gemm_bf16_nt<128,128,2,4><<<dim3(Sq / 128, Sq / 128, Bq * Hq), 256, 0, stream>>>(
        qkv_bf, qkv_bf + Dq, nullptr, scores, nullptr,
        Sq, Sq, HDq, 3 * Dq, 3 * Dq, Sq,
        Hq,
        (long long)Sq * 3 * Dq, HDq,
        (long long)Sq * 3 * Dq, HDq,
        (long long)Hq * Sq * Sq, (long long)Sq * Sq,
        iscale, 0);

    // softmax rows -> bf16 probs
    softmax_rows<<<Bq * Hq * Sq, 128, 0, stream>>>(scores, probs_bf);

    // O = P @ V   per (b,h): 512x64, K=512; scatter into (B,S,D) bf16
    gemm_bf16_nt<128,64,2,2><<<dim3(HDq / 64, Sq / 128, Bq * Hq), 256, 0, stream>>>(
        probs_bf, vt_bf, nullptr, nullptr, attn_bf,
        Sq, HDq, Sq, Sq, Sq, Dq,
        Hq,
        (long long)Hq * Sq * Sq, (long long)Sq * Sq,
        (long long)Hq * HDq * Sq, (long long)HDq * Sq,
        (long long)Sq * Dq, HDq,
        1.0f, 0);

    // attn proj: 4096x512, K=512 -> fp32 tmp
    gemm_bf16_nt<128,128,2,4><<<dim3(Dq / 128, BSq / 128, 1), 256, 0, stream>>>(
        attn_bf, w_outw + (size_t)i * Dq * Dq, out_b + (size_t)i * Dq,
        tmp, nullptr, BSq, Dq, Dq, Dq, Dq, Dq,
        1, 0, 0, 0, 0, 0, 0, 1.0f, 0);

    // h = LN(h + attn)
    ln_residual<<<BSq, 128, 0, stream>>>(h, tmp, ln1_g + (size_t)i * Dq,
                                         ln1_b + (size_t)i * Dq, h, hb);

    // ff1 = relu(h @ W1^T + b1)   4096x2048, K=512 -> bf16
    gemm_bf16_nt<128,128,2,4><<<dim3(Fq / 128, BSq / 128, 1), 256, 0, stream>>>(
        hb, w_ff1 + (size_t)i * Fq * Dq, ff1_b + (size_t)i * Fq,
        nullptr, ff1_bf, BSq, Fq, Dq, Dq, Dq, Fq,
        1, 0, 0, 0, 0, 0, 0, 1.0f, 1);

    // ff2 = ff1 @ W2^T + b2       4096x512, K=2048 -> fp32 tmp
    gemm_bf16_nt<128,128,2,4><<<dim3(Dq / 128, BSq / 128, 1), 256, 0, stream>>>(
        ff1_bf, w_ff2 + (size_t)i * Dq * Fq, ff2_b + (size_t)i * Dq,
        tmp, nullptr, BSq, Dq, Fq, Fq, Fq, Dq,
        1, 0, 0, 0, 0, 0, 0, 1.0f, 0);

    // h = LN(h + ff)
    ln_residual<<<BSq, 128, 0, stream>>>(h, tmp, ln2_g + (size_t)i * Dq,
                                         ln2_b + (size_t)i * Dq, h, hb);
  }

  // ---- six triad cycles
  for (int c = 0; c < Lq; ++c) {
    zero_f32<<<1, 256, 0, stream>>>(sqn, 1);
    sqnorm_partial<<<256, 256, 0, stream>>>(h, sqn);
    row_g_kernel<<<BSq, 128, 0, stream>>>(h, sqn, gbuf);
    quantile_mod<<<1, 1024, 0, stream>>>(gbuf, modrow, damp);
    triad_add<<<(BSq * Dq) / 256, 256, 0, stream>>>(h, hb, modrow);
  }

  // ---- logits = h @ Wout^T + bout   (4096 x 32000, K=512)
  gemm_bf16_nt<128,128,2,4><<<dim3(Vq / 128, BSq / 128, 1), 256, 0, stream>>>(
      hb, w_vout, bout, logits, nullptr,
      BSq, Vq, Dq, Dq, Dq, Vq,
      1, 0, 0, 0, 0, 0, 0, 1.0f, 0);
}
